// MultiHeadSyntonicAttention_1735166787890
// MI455X (gfx1250) — matrix operand order repaired and checked by reading, not compiled, not run on hardware
//
#include <hip/hip_runtime.h>

#define B_  2
#define S_  2048
#define D_  1024
#define H_  16
#define DH_ 64

typedef __bf16 bf16t;
typedef __attribute__((ext_vector_type(16))) __bf16 v16bf;
typedef __attribute__((ext_vector_type(8)))  __bf16 v8bf;
typedef __attribute__((ext_vector_type(8)))  float  v8f;
typedef unsigned int v4ui __attribute__((ext_vector_type(4)));
typedef int          v8i  __attribute__((ext_vector_type(8)));
typedef int          v4i  __attribute__((ext_vector_type(4)));

#if __has_include(<hip/amd_detail/amd_gfx1250_TDM.h>)
#define TDM_6ARG 1
#endif

#define CAT16(lo,hi) __builtin_shufflevector(lo, hi, 0,1,2,3,4,5,6,7,8,9,10,11,12,13,14,15)

__device__ __forceinline__ bf16t f2bf(float f) {
  unsigned u = __builtin_bit_cast(unsigned, f);
  u += 0x7FFFu + ((u >> 16) & 1u);              // round-to-nearest-even
  unsigned short h = (unsigned short)(u >> 16);
  return __builtin_bit_cast(bf16t, h);
}
__device__ __forceinline__ float bf2f(bf16t b) {
  unsigned short h = __builtin_bit_cast(unsigned short, b);
  unsigned u = ((unsigned)h) << 16;
  return __builtin_bit_cast(float, u);
}

// LDS byte offset of a __shared__ object (flat LDS pointers carry the
// wave-relative offset in the low 32 bits; aperture lives in the high bits).
__device__ __forceinline__ unsigned lds_off(const void* p) {
  return (unsigned)(unsigned long long)(uintptr_t)p;
}

// ---------------------------------------------------------------- TDM
// 2D tile DMA: tile_h rows x tile_w bf16 elements from a row-major matrix
// with row pitch `stride_elems`, into contiguous LDS at lds_byte.
// D# layout per CDNA5 ISA (S8.3/8.4):
//   g0: count=1 | lds_addr | global_addr[56:0] | type=2
//   g1: data_size=1(2B), tensor_dim0/1 = tile dims (tile always in-bounds),
//       tile_dim0/1, tensor_dim0_stride = row pitch
__device__ __forceinline__ void tdm_load_2d(unsigned lds_byte, const void* gptr,
                                            unsigned tile_w, unsigned tile_h,
                                            unsigned stride_elems)
{
  unsigned long long ga = (unsigned long long)(uintptr_t)gptr;
  v4ui g0;
  g0[0] = 1u;                                     // count=1, user mode
  g0[1] = lds_byte;                               // lds_addr
  g0[2] = (unsigned)ga;                           // global_addr[31:0]
  g0[3] = (unsigned)(ga >> 32) | (2u << 30);      // global_addr[56:32] | type=2
  v8i g1;
  g1[0] = (int)(1u << 16);                        // data_size = 2 bytes
  g1[1] = (int)((tile_w & 0xFFFFu) << 16);        // tensor_dim0[15:0]
  g1[2] = (int)((tile_w >> 16) | ((tile_h & 0xFFFFu) << 16)); // dim0 hi | dim1 lo
  g1[3] = (int)((tile_h >> 16) | ((tile_w & 0xFFFFu) << 16)); // dim1 hi | tile_dim0
  g1[4] = (int)(tile_h & 0xFFFFu);                // tile_dim1 (tile_dim2=0)
  g1[5] = (int)stride_elems;                      // tensor_dim0_stride[31:0]
  g1[6] = 0;                                      // stride hi | dim1_stride lo
  g1[7] = 0;
  v4i z = {0, 0, 0, 0};
#ifdef TDM_6ARG
  v8i z8 = {0, 0, 0, 0, 0, 0, 0, 0};
  __builtin_amdgcn_tensor_load_to_lds(g0, g1, z, z, z8, 0);
#else
  __builtin_amdgcn_tensor_load_to_lds(g0, g1, z, z, 0);
#endif
}

// ---------------------------------------------------------------- converters
__global__ __launch_bounds__(256)
void cvt_kernel(const float* __restrict__ in, bf16t* __restrict__ out, int n) {
  int i = blockIdx.x * 256 + threadIdx.x;
  if (i < n) out[i] = f2bf(in[i]);
}

// W is [K=1024][N=1024] row-major f32 -> Wt [N][K] bf16 (N-major for B frags)
__global__ __launch_bounds__(256)
void wtrans_kernel(const float* __restrict__ W, bf16t* __restrict__ Wt) {
  int i = blockIdx.x * 256 + threadIdx.x;   // i = n*1024 + k
  int n = i >> 10;
  int k = i & 1023;
  Wt[i] = f2bf(W[k * D_ + n]);
}

// ---------------------------------------------------------------- GEMM
// C[M,N] = A[M,K] @ B[K,N] + bias,  A row-major bf16, Bt = B^T (N-major) bf16.
// TDM double-buffered: wave 0 DMAs tile i+1 while all waves run WMMA on tile i.
// OUT_MODE 0: f32 C[M,N]   1: bf16 C[M,N]   2: bf16 transposed C[N,M]
template<int OUT_MODE>
__global__ __launch_bounds__(256)
void gemm_bf16_kernel(const bf16t* __restrict__ A, const bf16t* __restrict__ Bt,
                      const float* __restrict__ bias, void* __restrict__ Cout,
                      int Mtot, int Ntot, int K)
{
  __shared__ bf16t As[2][128 * 32];
  __shared__ bf16t Bs[2][128 * 32];

  const int tid  = threadIdx.x;
  const int lane = tid & 31;
  const int wave = tid >> 5;
  const int wm   = wave >> 2;           // 0..1  (64 rows each)
  const int wn   = wave & 3;            // 0..3  (32 cols each)
  const int mBase = blockIdx.y * 128;
  const int nBase = blockIdx.x * 128;

  const int half8  = (lane >> 4) ? 8  : 0;   // A-frag K offset (interleaved)
  const int half16 = (lane >> 4) ? 16 : 0;   // unused: B frags also take half8
  const int mrow8  = (lane >> 4) << 3;

  v8f acc[4][2] = {};

  const int nsteps = K >> 5;
  if (wave == 0) {                      // prologue DMA: tile 0
    tdm_load_2d(lds_off(&As[0][0]), &A [(size_t)mBase * K], 32, 128, (unsigned)K);
    tdm_load_2d(lds_off(&Bs[0][0]), &Bt[(size_t)nBase * K], 32, 128, (unsigned)K);
  }

  for (int i = 0; i < nsteps; ++i) {
    const int buf = i & 1;
    if (wave == 0) {
      if (i + 1 < nsteps) {             // DMA next tile into the other buffer
        tdm_load_2d(lds_off(&As[buf ^ 1][0]),
                    &A [(size_t)mBase * K + (size_t)(i + 1) * 32], 32, 128, (unsigned)K);
        tdm_load_2d(lds_off(&Bs[buf ^ 1][0]),
                    &Bt[(size_t)nBase * K + (size_t)(i + 1) * 32], 32, 128, (unsigned)K);
        __builtin_amdgcn_s_wait_tensorcnt(2);  // in-order: tile i resident
      } else {
        __builtin_amdgcn_s_wait_tensorcnt(0);
      }
    }
    __syncthreads();

    v16bf afr[4], bfr[2];
    #pragma unroll
    for (int mt = 0; mt < 4; ++mt) {
      int r = wm * 64 + mt * 16 + (lane & 15);
      v8bf lo = *(const v8bf*)&As[buf][r * 32 + half8];
      v8bf hi = *(const v8bf*)&As[buf][r * 32 + half8 + 16];
      afr[mt] = CAT16(lo, hi);
    }
    #pragma unroll
    for (int nt = 0; nt < 2; ++nt) {
      int r = wn * 32 + nt * 16 + (lane & 15);
      v8bf lo = *(const v8bf*)&Bs[buf][r * 32 + half8];
      v8bf hi = *(const v8bf*)&Bs[buf][r * 32 + half8 + 16];
      bfr[nt] = CAT16(lo, hi);
    }
    #pragma unroll
    for (int mt = 0; mt < 4; ++mt)
      #pragma unroll
      for (int nt = 0; nt < 2; ++nt)
        acc[mt][nt] = __builtin_amdgcn_wmma_f32_16x16x32_bf16(
            false, afr[mt], false, bfr[nt], (short)0, acc[mt][nt], false, false);
    __syncthreads();                    // waves done with buf before it is re-filled
  }

  #pragma unroll
  for (int mt = 0; mt < 4; ++mt) {
    int m0 = mBase + wm * 64 + mt * 16 + mrow8;
    #pragma unroll
    for (int nt = 0; nt < 2; ++nt) {
      int n = nBase + wn * 32 + nt * 16 + (lane & 15);
      float bv = bias ? bias[n] : 0.f;
      if (OUT_MODE == 0) {
        float* C = (float*)Cout;
        #pragma unroll
        for (int j = 0; j < 8; ++j) C[(size_t)(m0 + j) * Ntot + n] = acc[mt][nt][j] + bv;
      } else if (OUT_MODE == 1) {
        bf16t* C = (bf16t*)Cout;
        #pragma unroll
        for (int j = 0; j < 8; ++j) C[(size_t)(m0 + j) * Ntot + n] = f2bf(acc[mt][nt][j] + bv);
      } else {
        bf16t* C = (bf16t*)Cout;               // [Ntot][Mtot]
        v8bf pk;
        #pragma unroll
        for (int j = 0; j < 8; ++j) pk[j] = f2bf(acc[mt][nt][j] + bv);
        *(v8bf*)&C[(size_t)n * Mtot + m0] = pk;
      }
    }
  }
}

// ---------------------------------------------------------------- gating
// gn[b*H+h][s] = sigmoid( K[b,s,h*64:..] . wg + bg )
__global__ __launch_bounds__(256)
void gn_kernel(const bf16t* __restrict__ Kb, const float* __restrict__ wg,
               const float* __restrict__ bg, float* __restrict__ gn)
{
  int idx = blockIdx.x * 256 + threadIdx.x;   // < B*H*S
  int s   = idx & (S_ - 1);
  int bh  = idx >> 11;                        // /2048
  int b   = bh >> 4, h = bh & 15;
  const bf16t* kp = &Kb[((size_t)b * S_ + s) * D_ + h * DH_];
  float sum = 0.f;
  #pragma unroll 8
  for (int c = 0; c < DH_; ++c) sum += bf2f(kp[c]) * wg[c];
  sum += bg[0];
  gn[idx] = 1.f / (1.f + __expf(-sum));
}

// ---------------------------------------------------------------- attention
// Per (b,h, 64 query rows): flash-style over 64-key chunks; Q/K/V tiles DMAed
// into LDS by the TDM, scores + AV on WMMA, online softmax in between.
__global__ __launch_bounds__(256)
void attn_kernel(const bf16t* __restrict__ Qb, const bf16t* __restrict__ Kb,
                 const bf16t* __restrict__ Vt, const float* __restrict__ gn,
                 bf16t* __restrict__ Out)
{
  __shared__ bf16t Qs[64 * 64];
  __shared__ bf16t Ks[64 * 64];
  __shared__ bf16t Vs[64 * 64];     // [dh][kpos]
  __shared__ float Ss[64 * 68];     // padded
  __shared__ bf16t Ps[64 * 64];
  __shared__ float rowA[64];
  __shared__ float rowL[64];

  const int tid  = threadIdx.x;
  const int lane = tid & 31;
  const int wave = tid >> 5;
  const int wm   = wave >> 2;       // 0..1 (32 rows each)
  const int wn   = wave & 3;        // 0..3 (16 cols each)
  const int bh   = blockIdx.x;
  const int b    = bh >> 4, h = bh & 15;
  const int qbase = blockIdx.y * 64;
  const size_t BSl = (size_t)B_ * S_;

  const int half8  = (lane >> 4) ? 8  : 0;
  const int half16 = (lane >> 4) ? 16 : 0;
  const int mrow8  = (lane >> 4) << 3;
  const int nloc   = wn * 16 + (lane & 15);

  if (wave == 0) {                  // Q tile: 64 x 64 bf16, row pitch D
    tdm_load_2d(lds_off(Qs), &Qb[((size_t)b * S_ + qbase) * D_ + h * DH_],
                64, 64, D_);
  }

  v8f oacc[2] = {};
  float m_i = -__builtin_inff(), l_i = 0.f;
  const float scale = 0.125f;       // 1/sqrt(64)

  for (int kk = 0; kk < S_; kk += 64) {
    __syncthreads();                // everyone done with previous K/V/P tiles
    if (wave == 0) {
      tdm_load_2d(lds_off(Ks), &Kb[((size_t)b * S_ + kk) * D_ + h * DH_],
                  64, 64, D_);
      tdm_load_2d(lds_off(Vs), &Vt[(size_t)(h * DH_) * BSl + (size_t)b * S_ + kk],
                  64, 64, (unsigned)BSl);
      __builtin_amdgcn_s_wait_tensorcnt(0);   // also covers the Q tile
    }
    __syncthreads();

    // scores S = Q @ K^T  (K-dim = dh = 64 -> 2 wmma steps)
    v8f sacc[2] = {};
    #pragma unroll
    for (int ks = 0; ks < 2; ++ks) {
      v8bf blo = *(const v8bf*)&Ks[nloc * 64 + ks * 32 + half8];
      v8bf bhi = *(const v8bf*)&Ks[nloc * 64 + ks * 32 + half8 + 16];
      v16bf bfr = CAT16(blo, bhi);
      #pragma unroll
      for (int mt = 0; mt < 2; ++mt) {
        int r = wm * 32 + mt * 16 + (lane & 15);
        v8bf alo = *(const v8bf*)&Qs[r * 64 + ks * 32 + half8];
        v8bf ahi = *(const v8bf*)&Qs[r * 64 + ks * 32 + half8 + 16];
        v16bf afr = CAT16(alo, ahi);
        sacc[mt] = __builtin_amdgcn_wmma_f32_16x16x32_bf16(
            false, afr, false, bfr, (short)0, sacc[mt], false, false);
      }
    }
    float g = scale * (1.f + gn[(size_t)bh * S_ + kk + nloc]);  // per key column
    #pragma unroll
    for (int mt = 0; mt < 2; ++mt)
      #pragma unroll
      for (int j = 0; j < 8; ++j)
        Ss[(wm * 32 + mt * 16 + mrow8 + j) * 68 + nloc] = sacc[mt][j] * g;
    __syncthreads();

    // online softmax: one thread per query row
    if (tid < 64) {
      int r = tid;
      float mx = -__builtin_inff();
      for (int k2 = 0; k2 < 64; ++k2) mx = fmaxf(mx, Ss[r * 68 + k2]);
      float mnew  = fmaxf(m_i, mx);
      float alpha = __expf(m_i - mnew);
      float sum = 0.f;
      for (int k2 = 0; k2 < 64; ++k2) {
        float e = __expf(Ss[r * 68 + k2] - mnew);
        Ps[r * 64 + k2] = f2bf(e);
        sum += e;
      }
      l_i = l_i * alpha + sum;
      m_i = mnew;
      rowA[r] = alpha;
      rowL[r] = l_i;
    }
    __syncthreads();

    // O = O*alpha + P @ V
    #pragma unroll
    for (int mt = 0; mt < 2; ++mt)
      #pragma unroll
      for (int j = 0; j < 8; ++j)
        oacc[mt][j] *= rowA[wm * 32 + mt * 16 + mrow8 + j];
    #pragma unroll
    for (int ks = 0; ks < 2; ++ks) {
      v8bf blo = *(const v8bf*)&Vs[nloc * 64 + ks * 32 + half8];
      v8bf bhi = *(const v8bf*)&Vs[nloc * 64 + ks * 32 + half8 + 16];
      v16bf bfr = CAT16(blo, bhi);
      #pragma unroll
      for (int mt = 0; mt < 2; ++mt) {
        int r = wm * 32 + mt * 16 + (lane & 15);
        v8bf alo = *(const v8bf*)&Ps[r * 64 + ks * 32 + half8];
        v8bf ahi = *(const v8bf*)&Ps[r * 64 + ks * 32 + half8 + 16];
        v16bf afr = CAT16(alo, ahi);
        oacc[mt] = __builtin_amdgcn_wmma_f32_16x16x32_bf16(
            false, afr, false, bfr, (short)0, oacc[mt], false, false);
      }
    }
  }
  __syncthreads();

  #pragma unroll
  for (int mt = 0; mt < 2; ++mt)
    #pragma unroll
    for (int j = 0; j < 8; ++j) {
      int m = wm * 32 + mt * 16 + mrow8 + j;
      float inv = 1.f / rowL[m];
      Out[((size_t)b * S_ + qbase + m) * D_ + h * DH_ + nloc] = f2bf(oacc[mt][j] * inv);
    }
}

// ---------------------------------------------------------------- launch
extern "C" void kernel_launch(void* const* d_in, const int* in_sizes, int n_in,
                              void* d_out, int out_size, void* d_ws, size_t ws_size,
                              hipStream_t stream)
{
  (void)in_sizes; (void)n_in; (void)out_size; (void)ws_size;
  const float* query = (const float*)d_in[0];
  const float* key_  = (const float*)d_in[1];
  const float* value = (const float*)d_in[2];
  const float* Wq = (const float*)d_in[3];  const float* bq = (const float*)d_in[4];
  const float* Wk = (const float*)d_in[5];  const float* bk = (const float*)d_in[6];
  const float* Wv = (const float*)d_in[7];  const float* bv = (const float*)d_in[8];
  const float* wg = (const float*)d_in[9];  const float* bg = (const float*)d_in[10];
  const float* Wo = (const float*)d_in[11]; const float* bo = (const float*)d_in[12];
  const float* Wd = (const float*)d_in[13]; const float* bd = (const float*)d_in[14];
  const float* Wh = (const float*)d_in[15]; const float* bh = (const float*)d_in[16];

  const int    MS   = B_ * S_;                  // 4096 rows
  const size_t actB = (size_t)MS * D_ * sizeof(bf16t);   // 8 MB
  const size_t wB   = (size_t)D_ * D_ * sizeof(bf16t);   // 2 MB

  char* p = (char*)d_ws; size_t off = 0;
  auto take = [&](size_t bytes) -> void* {
    void* r = p + off; off += (bytes + 255) & ~(size_t)255; return r;
  };
  bf16t* Xq  = (bf16t*)take(actB);
  bf16t* Xk  = (bf16t*)take(actB);
  bf16t* Xv  = (bf16t*)take(actB);
  bf16t* Wqt = (bf16t*)take(wB);
  bf16t* Wkt = (bf16t*)take(wB);
  bf16t* Wvt = (bf16t*)take(wB);
  bf16t* Wot = (bf16t*)take(wB);
  bf16t* Wdt = (bf16t*)take(wB);
  bf16t* Wht = (bf16t*)take(wB);
  bf16t* Qbf = (bf16t*)take(actB);
  bf16t* Kbf = (bf16t*)take(actB);
  bf16t* Vtb = (bf16t*)take(actB);              // transposed [D][B*S]
  float* gnb = (float*)take((size_t)B_ * H_ * S_ * sizeof(float));
  bf16t* att = (bf16t*)take(actB);
  bf16t* t1  = (bf16t*)take(actB);
  bf16t* t2  = (bf16t*)take(actB);
  // total ~84 MB of d_ws

  const int nAct = MS * D_;
  cvt_kernel<<<(nAct + 255) / 256, 256, 0, stream>>>(query, Xq, nAct);
  cvt_kernel<<<(nAct + 255) / 256, 256, 0, stream>>>(key_,  Xk, nAct);
  cvt_kernel<<<(nAct + 255) / 256, 256, 0, stream>>>(value, Xv, nAct);
  const int nW = D_ * D_;
  wtrans_kernel<<<nW / 256, 256, 0, stream>>>(Wq, Wqt);
  wtrans_kernel<<<nW / 256, 256, 0, stream>>>(Wk, Wkt);
  wtrans_kernel<<<nW / 256, 256, 0, stream>>>(Wv, Wvt);
  wtrans_kernel<<<nW / 256, 256, 0, stream>>>(Wo, Wot);
  wtrans_kernel<<<nW / 256, 256, 0, stream>>>(Wd, Wdt);
  wtrans_kernel<<<nW / 256, 256, 0, stream>>>(Wh, Wht);

  dim3 gg(D_ / 128, MS / 128);                  // (8, 32)
  gemm_bf16_kernel<1><<<gg, 256, 0, stream>>>(Xq, Wqt, bq, Qbf, MS, D_, D_);
  gemm_bf16_kernel<1><<<gg, 256, 0, stream>>>(Xk, Wkt, bk, Kbf, MS, D_, D_);
  gemm_bf16_kernel<2><<<gg, 256, 0, stream>>>(Xv, Wvt, bv, Vtb, MS, D_, D_);

  gn_kernel<<<(B_ * H_ * S_) / 256, 256, 0, stream>>>(Kbf, wg, bg, gnb);

  attn_kernel<<<dim3(B_ * H_, S_ / 64), 256, 0, stream>>>(Qbf, Kbf, Vtb, gnb, att);

  gemm_bf16_kernel<1><<<gg, 256, 0, stream>>>(att, Wot, bo, t1, MS, D_, D_);
  gemm_bf16_kernel<1><<<gg, 256, 0, stream>>>(t1,  Wdt, bd, t2, MS, D_, D_);
  gemm_bf16_kernel<0><<<gg, 256, 0, stream>>>(t2,  Wht, bh, d_out, MS, D_, D_);
}
